// HEALUpSampler_40518721470592
// MI455X (gfx1250) — matrix operand order repaired and checked by reading, not compiled, not run on hardware
//
#include <hip/hip_runtime.h>

typedef __attribute__((ext_vector_type(16))) __bf16 v16bf;
typedef __attribute__((ext_vector_type(8)))  __bf16 v8bf;
typedef __attribute__((ext_vector_type(8)))  float  v8f;

#define N_REC   196608
#define D_H     256
#define MROWS   64            // receiver rows per workgroup (4 waves x 16)
#define LDS_STRIDE_F 132      // f32 stride, padded (528B, 16B aligned)
#define LDS_STRIDE_H 264      // same bytes as bf16 stride

__device__ __forceinline__ float gelu_tanh(float x) {
    const float k0 = 0.7978845608028654f;  // sqrt(2/pi)
    const float k1 = 0.044715f;
    float x3 = x * x * x;
    return 0.5f * x * (1.0f + tanhf(k0 * (x + k1 * x3)));
}

// ---------------------------------------------------------------------------
// efsum[d] = sum_{k=0..3} ( gelu(edge_attr[k]*We1 + be1) @ We2 + be2 )[d]
// (every receiver's 4 edges carry attrs edge_attr[0..3] = 0,1,2,3)
// ---------------------------------------------------------------------------
__global__ __launch_bounds__(128) void prep_ef(const float* __restrict__ edge_attr,
                                               const float* __restrict__ We1,
                                               const float* __restrict__ be1,
                                               const float* __restrict__ We2,
                                               const float* __restrict__ be2,
                                               float* __restrict__ efsum) {
    __shared__ float hsum[128];
    int j = threadIdx.x;
    float w1 = We1[j], b1 = be1[j];
    float s = 0.0f;
    #pragma unroll
    for (int k = 0; k < 4; ++k) s += gelu_tanh(edge_attr[k] * w1 + b1);
    hsum[j] = s;
    __syncthreads();
    float acc = 4.0f * be2[j];
    for (int jj = 0; jj < 128; ++jj) acc += hsum[jj] * We2[jj * 128 + j];
    efsum[j] = acc;
}

// ---------------------------------------------------------------------------
// Repack Wl1/Wl2 (256x256 f32 row-major [k][n]) into bf16 B-fragment order:
// frag (ktile,ntile): lane holds B[ktile*32 + (lane>>4)*8 + i + (i>=8 ? 8 : 0)][ntile*16 + (lane&15)]
// stored contiguously so each lane does one 32B load per fragment.
// ---------------------------------------------------------------------------
__global__ __launch_bounds__(32) void prep_weights(const float* __restrict__ Wl1,
                                                   const float* __restrict__ Wl2,
                                                   __bf16* __restrict__ wB1,
                                                   __bf16* __restrict__ wB2) {
    int bid  = blockIdx.x;           // 256 blocks: mat(1) x ktile(3) x ntile(4)
    int mat  = bid >> 7;
    int kt   = (bid >> 4) & 7;
    int nt   = bid & 15;
    int lane = threadIdx.x;
    const float* W = mat ? Wl2 : Wl1;
    __bf16* dst = (mat ? wB2 : wB1) + ((kt * 16 + nt) * 512 + lane * 16);
    int krow = kt * 32 + (lane >> 4) * 8;
    int n    = nt * 16 + (lane & 15);
    #pragma unroll
    for (int i = 0; i < 16; ++i) dst[i] = (__bf16)W[(krow + i + (i >= 8 ? 8 : 0)) * 256 + n];
}

// ---------------------------------------------------------------------------
// Fused: gather-sum -> GEMM1(bf16 WMMA) -> gelu -> GEMM2 -> +bias -> out
// ---------------------------------------------------------------------------
__global__ __launch_bounds__(128) void fused_main(const float* __restrict__ x,
                                                  const int*   __restrict__ senders,
                                                  const float* __restrict__ efsum,
                                                  const __bf16* __restrict__ wB1,
                                                  const __bf16* __restrict__ wB2,
                                                  const float* __restrict__ bl1,
                                                  const float* __restrict__ bl2,
                                                  float* __restrict__ out) {
    __shared__ float lds[MROWS * LDS_STRIDE_F];   // 33,792 B; reused as bf16 h
    const int tid      = threadIdx.x;
    const int lane     = tid & 31;
    const int wave     = tid >> 5;
    const int blockRow = blockIdx.x * MROWS;

    // ---- Phase 1: agg[row][0:128] = sum of 4 gathered x rows (coalesced) ----
    #pragma unroll
    for (int it = 0; it < 16; ++it) {
        int flat = it * 128 + tid;          // 64 rows x 32 float4-groups
        int row  = flat >> 5;
        int cg   = flat & 31;
        const int* s4 = senders + 4 * (blockRow + row);
        float4 acc = make_float4(0.f, 0.f, 0.f, 0.f);
        #pragma unroll
        for (int k = 0; k < 4; ++k) {
            const float4 v = *(const float4*)(x + (long)s4[k] * 128 + cg * 4);
            acc.x += v.x; acc.y += v.y; acc.z += v.z; acc.w += v.w;
        }
        *(float4*)(lds + row * LDS_STRIDE_F + cg * 4) = acc;
    }
    __syncthreads();

    // ---- Phase 2: per-wave 16 rows, full N=256 ----
    const int mrow  = wave * 16;
    const int mlane = lane & 15;
    const int khalf = lane >> 4;            // selects K sub-groups per ISA A layout

    // A fragments, layer 1 (K=256): ktiles 0..3 from gathered agg, 4..7 = efsum
    v16bf a[8];
    #pragma unroll
    for (int kt = 0; kt < 4; ++kt) {
        int kb = kt * 32 + khalf * 8;
        const float* src = lds + (mrow + mlane) * LDS_STRIDE_F + kb;
        #pragma unroll
        for (int i = 0; i < 8; ++i) {
            a[kt][i]     = (__bf16)src[i];        // K = kb + i
            a[kt][8 + i] = (__bf16)src[16 + i];   // K = kb + 16 + i
        }
    }
    #pragma unroll
    for (int kt = 4; kt < 8; ++kt) {
        int kb = (kt - 4) * 32 + khalf * 8;
        #pragma unroll
        for (int i = 0; i < 8; ++i) {
            a[kt][i]     = (__bf16)efsum[kb + i];
            a[kt][8 + i] = (__bf16)efsum[kb + 16 + i];
        }
    }

    // ---- GEMM1 ----
    v8f acc[16];
    #pragma unroll
    for (int nt = 0; nt < 16; ++nt) acc[nt] = v8f{0.f,0.f,0.f,0.f,0.f,0.f,0.f,0.f};
    #pragma unroll
    for (int kt = 0; kt < 8; ++kt) {
        #pragma unroll
        for (int nt = 0; nt < 16; ++nt) {
            v16bf b = *(const v16bf*)(wB1 + ((kt * 16 + nt) * 512 + lane * 16));
            acc[nt] = __builtin_amdgcn_wmma_f32_16x16x32_bf16(
                false, a[kt], false, b, (short)0, acc[nt], false, false);
        }
    }

    // ---- gelu(acc + bl1) -> h staged in LDS as bf16 (wave-private rows) ----
    __bf16* hLds = (__bf16*)lds;
    #pragma unroll
    for (int nt = 0; nt < 16; ++nt) {
        int n = nt * 16 + mlane;
        float b1 = bl1[n];
        #pragma unroll
        for (int j = 0; j < 8; ++j) {
            int rloc = mrow + j + khalf * 8;    // C layout: VGPR j -> M = j (+8)
            hLds[rloc * LDS_STRIDE_H + n] = (__bf16)gelu_tanh(acc[nt][j] + b1);
        }
    }

    // A fragments, layer 2 (from bf16 h, aligned 16B ds loads)
    v16bf a2[8];
    #pragma unroll
    for (int kt = 0; kt < 8; ++kt) {
        int kb = kt * 32 + khalf * 8;
        const __bf16* src = hLds + (mrow + mlane) * LDS_STRIDE_H + kb;
        v8bf lo = *(const v8bf*)(src);
        v8bf hi = *(const v8bf*)(src + 16);
        a2[kt] = __builtin_shufflevector(lo, hi, 0,1,2,3,4,5,6,7,8,9,10,11,12,13,14,15);
    }

    // ---- GEMM2 ----
    #pragma unroll
    for (int nt = 0; nt < 16; ++nt) acc[nt] = v8f{0.f,0.f,0.f,0.f,0.f,0.f,0.f,0.f};
    #pragma unroll
    for (int kt = 0; kt < 8; ++kt) {
        #pragma unroll
        for (int nt = 0; nt < 16; ++nt) {
            v16bf b = *(const v16bf*)(wB2 + ((kt * 16 + nt) * 512 + lane * 16));
            acc[nt] = __builtin_amdgcn_wmma_f32_16x16x32_bf16(
                false, a2[kt], false, b, (short)0, acc[nt], false, false);
        }
    }

    // ---- epilogue: out = acc + bl2 (f32) ----
    #pragma unroll
    for (int nt = 0; nt < 16; ++nt) {
        int n = nt * 16 + mlane;
        float b2 = bl2[n];
        #pragma unroll
        for (int j = 0; j < 8; ++j) {
            int grow = blockRow + mrow + j + khalf * 8;
            out[(long)grow * D_H + n] = acc[nt][j] + b2;
        }
    }
}

extern "C" void kernel_launch(void* const* d_in, const int* in_sizes, int n_in,
                              void* d_out, int out_size, void* d_ws, size_t ws_size,
                              hipStream_t stream) {
    const float* x          = (const float*)d_in[0];
    const int*   edge_index = (const int*)d_in[1];   // [0..E): senders
    const float* edge_attr  = (const float*)d_in[2];
    const float* We1        = (const float*)d_in[3];
    const float* be1        = (const float*)d_in[4];
    const float* We2        = (const float*)d_in[5];
    const float* be2        = (const float*)d_in[6];
    const float* Wl1        = (const float*)d_in[7];
    const float* bl1        = (const float*)d_in[8];
    const float* Wl2        = (const float*)d_in[9];
    const float* bl2        = (const float*)d_in[10];

    float*  efsum = (float*)d_ws;                               // 512 B
    __bf16* wB1   = (__bf16*)((char*)d_ws + 512);               // 128 KB
    __bf16* wB2   = (__bf16*)((char*)d_ws + 512 + 131072);      // 128 KB

    prep_ef<<<1, 128, 0, stream>>>(edge_attr, We1, be1, We2, be2, efsum);
    prep_weights<<<256, 32, 0, stream>>>(Wl1, Wl2, wB1, wB2);
    fused_main<<<N_REC / MROWS, 128, 0, stream>>>(x, edge_index, efsum, wB1, wB2,
                                                  bl1, bl2, (float*)d_out);
}
